// BasicClassifier_46737834115479
// MI455X (gfx1250) — matrix operand order repaired and checked by reading, not compiled, not run on hardware
//
#include <hip/hip_runtime.h>

// ---------------------------------------------------------------------------
// Fused SNN forward for MI455X (gfx1250, wave32, WMMA 16x16x32 f16 -> f32).
// One workgroup = 16 batch elements (one WMMA M-tile), 8 waves, loops all
// T=100 timesteps. Spikes pass between layers through LDS as f16 (exact 0/1).
// conv2 + conv3 membranes live in LDS as lane-private packed 8-float chunks
// (tile->lane ownership fixed across time, so no synchronization needed);
// the small conv1/fc membranes stay in VGPRs. Weights are pre-swizzled into
// per-lane WMMA B-fragment order by prep kernels so the hot loop does only
// contiguous b128 loads. conv2/conv3 tile loops are kept rolled to bound
// register pressure under 256 VGPRs (no MSB banking, no spills).
// ---------------------------------------------------------------------------

typedef _Float16 v8h  __attribute__((ext_vector_type(8)));
typedef _Float16 v16h __attribute__((ext_vector_type(16)));
typedef float    v8f  __attribute__((ext_vector_type(8)));

union AFrag { v16h v; v8h h[2]; };
union MChunk { float4 v[2]; float f[8]; };

#define WMMA_F16(a, b, c) \
  __builtin_amdgcn_wmma_f32_16x16x32_f16(false, (a), false, (b), (short)0, (c), false, false)

// workspace layout (units: _Float16 halves). One 16x16x32 fragment of B is
// 32 lanes * 16 halves = 512 halves. frag index = n_tile*KSTEPS + k_step.
constexpr int OFF_W2  = 0;              // conv2: 2 ntiles * 3 ks * 512 = 3072
constexpr int OFF_W3  = 3072;           // conv3: 4 ntiles * 3 ks * 512 = 6144
constexpr int OFF_FW1 = 9216;           // fc1 : 16 ntiles * 36 ks * 512 = 294912
constexpr int OFF_FW2 = 304128;         // fc2 : 1 ntile * 8 ks * 512 = 4096
// total = 308224 halves = 616448 bytes of d_ws

// ---------------------------------------------------------------------------
// Weight prepack kernels: produce f16 B-matrix fragments in lane order.
// ISA 16-bit B layout (K32 x N16): lane l<16 holds K=0..7,16..23 of column N=l,
// lane l>=16 holds K=8..15,24..31 of column N=l-16; 16 contiguous halves per lane.
// half address h = (frag*32 + lane)*16 + j ; K = ks*32 + 8*(l>>4) + 16*(j>>3) + (j&7).
// ---------------------------------------------------------------------------

__global__ void pack_fw1(const float* __restrict__ fw1, _Float16* __restrict__ ws) {
  int h    = blockIdx.x * 256 + threadIdx.x;       // 294912 halves
  int j    = h & 15;
  int lane = (h >> 4) & 31;
  int f    = h >> 9;                                // nt*36 + ks
  int ks   = f % 36, nt = f / 36;
  int K    = ks * 32 + (lane >> 4) * 8 + (j >> 3) * 16 + (j & 7);  // K over LDS layout pos*64+ch
  int N    = nt * 16 + (lane & 15);                 // 0..255
  int ch   = K & 63, pos = K >> 6;                  // invert reshape: f = ch*18+pos
  ws[OFF_FW1 + h] = (_Float16)fw1[N * 1152 + ch * 18 + pos];
}

__global__ void pack_w2(const float* __restrict__ w2, _Float16* __restrict__ ws) {
  int h    = blockIdx.x * 256 + threadIdx.x;       // 3072 halves
  int j    = h & 15;
  int lane = (h >> 4) & 31;
  int f    = h >> 9;                                // nt*3 + ks
  int ks   = f % 3, nt = f / 3;
  int K    = ks * 32 + (lane >> 4) * 8 + (j >> 3) * 16 + (j & 7);  // tap*16 + ci, padded to 96
  int N    = nt * 16 + (lane & 15);                 // co 0..31
  int tap  = K >> 4, ci = K & 15;
  float v  = (K < 80) ? w2[(N * 16 + ci) * 5 + tap] : 0.0f;  // zero phantom K
  ws[OFF_W2 + h] = (_Float16)v;
}

__global__ void pack_w3(const float* __restrict__ w3, _Float16* __restrict__ ws) {
  int h    = blockIdx.x * 256 + threadIdx.x;       // 6144 halves
  int j    = h & 15;
  int lane = (h >> 4) & 31;
  int f    = h >> 9;                                // nt*3 + ks
  int ks   = f % 3, nt = f / 3;
  int K    = ks * 32 + (lane >> 4) * 8 + (j >> 3) * 16 + (j & 7);  // tap*32 + ci, K=96 exact
  int N    = nt * 16 + (lane & 15);                 // co 0..63
  int tap  = K >> 5, ci = K & 31;
  ws[OFF_W3 + h] = (_Float16)w3[(N * 32 + ci) * 3 + tap];
}

__global__ void pack_fw2(const float* __restrict__ fw2, _Float16* __restrict__ ws) {
  int h    = blockIdx.x * 256 + threadIdx.x;       // 4096 halves
  int j    = h & 15;
  int lane = (h >> 4) & 31;
  int ks   = h >> 9;                                // 0..7
  int K    = ks * 32 + (lane >> 4) * 8 + (j >> 3) * 16 + (j & 7);  // 0..255
  int N    = lane & 15;                              // pad cols 10..15 -> 0
  float v  = (N < 10) ? fw2[N * 256 + K] : 0.0f;
  ws[OFF_FW2 + h] = (_Float16)v;
}

// ---------------------------------------------------------------------------
// Main fused kernel. grid = 64 blocks (B/16), 256 threads (8 wave32s).
// LIF (snntorch Leaky, subtract reset): rst=(m>1); m=0.9m+inp-rst; s=(m>1).
// ---------------------------------------------------------------------------
__launch_bounds__(256)
__global__ void snn_fused(const float* __restrict__ x,
                          const float* __restrict__ w1, const float* __restrict__ b1,
                          const float* __restrict__ b2, const float* __restrict__ b3,
                          const float* __restrict__ fb1, const float* __restrict__ fb2,
                          const _Float16* __restrict__ wpk,
                          float* __restrict__ out) {
  // spikes laid out [batch][pos][ch], ch contiguous (K-inner for WMMA A reads)
  __shared__ __align__(16) float    xs[16 * 30];          // raw input rows   (1.9 KB)
  __shared__ __align__(16) _Float16 s1[16 * 25 * 16];     // conv1 spikes     (12.5 KB)
  __shared__ __align__(16) _Float16 s2[16 * 20 * 32];     // conv2 spikes     (20 KB)
  __shared__ __align__(16) _Float16 s3[16 * 1152];        // conv3 spikes     (36 KB)
  __shared__ __align__(16) _Float16 s4[16 * 256];         // fc1 spikes       (8 KB)
  // conv membranes: lane-private packed chunks [tile][hi][lm][8 rows]
  __shared__ __align__(16) float    m2lds[40 * 2 * 16 * 8];   // 40 KB
  __shared__ __align__(16) float    m3lds[72 * 2 * 16 * 8];   // 72 KB
  __shared__ __align__(16) float    blds[96];             // staged b2(32) + b3(64)

  const int tid  = threadIdx.x;
  const int w    = tid >> 5;          // wave 0..7
  const int lane = tid & 31;
  const int lm   = lane & 15;         // N column / A batch row
  const int hi   = lane >> 4;         // lane half select
  const int bg   = blockIdx.x * 16;   // global batch base

  // zero spike + membrane LDS once (covers s1's phantom-tap pad rows)
  for (int i = tid; i < 16 * 25 * 16; i += 256) s1[i] = (_Float16)0.0f;
  for (int i = tid; i < 16 * 20 * 32; i += 256) s2[i] = (_Float16)0.0f;
  for (int i = tid; i < 16 * 1152;    i += 256) s3[i] = (_Float16)0.0f;
  for (int i = tid; i < 16 * 256;     i += 256) s4[i] = (_Float16)0.0f;
  for (int i = tid; i < 40 * 2 * 16 * 8; i += 256) m2lds[i] = 0.0f;
  for (int i = tid; i < 72 * 2 * 16 * 8; i += 256) m3lds[i] = 0.0f;
  if (tid < 32) blds[tid] = b2[tid];
  else if (tid < 96) blds[tid] = b3[tid - 32];

  // ---- conv1 per-thread setup: thread = (batch cb, channel cc) ----
  const int cb = tid >> 4;
  const int cc = tid & 15;
  float wr[7];
#pragma unroll
  for (int j = 0; j < 7; ++j) wr[j] = w1[cc * 7 + j];
  const float bc1 = b1[cc];

  // fc biases (static per wave)
  float bias4[2];
#pragma unroll
  for (int i = 0; i < 2; ++i) bias4[i] = fb1[(w * 2 + i) * 16 + lm];
  const float bias5 = (lm < 10) ? fb2[lm] : 0.0f;

  // ---- register-resident membranes (small ones only) ----
  float m1[24];                 // conv1: this thread's (cb,cc) row, 24 positions
  float m4[2][8];               // fc1  : 2 N-tiles per wave
  float m5[8];                  // fc2  : wave 0 only
#pragma unroll
  for (int p = 0; p < 24; ++p) m1[p] = 0.0f;
#pragma unroll
  for (int i = 0; i < 2; ++i)
#pragma unroll
    for (int r = 0; r < 8; ++r) m4[i][r] = 0.0f;
#pragma unroll
  for (int r = 0; r < 8; ++r) m5[r] = 0.0f;

  __syncthreads();

  for (int t = 0; t < 100; ++t) {
    // ---------- stage x[:, 0, :, t] for this batch tile ----------
    for (int idx = tid; idx < 480; idx += 256) {
      int bb = idx / 30, l = idx - bb * 30;
      xs[idx] = x[((bg + bb) * 30 + l) * 100 + t];   // x is (B,1,30,100)
    }
    __syncthreads();

    // ---------- conv1 (1->16, k=7) + LIF : pure VALU ----------
#pragma unroll
    for (int p = 0; p < 24; ++p) {
      float h = bc1;
#pragma unroll
      for (int j = 0; j < 7; ++j) h += wr[j] * xs[cb * 30 + p + j];
      float m   = m1[p];
      float rst = (m > 1.0f) ? 1.0f : 0.0f;
      m = 0.9f * m + h - rst;
      m1[p] = m;
      s1[(cb * 25 + p) * 16 + cc] = (m > 1.0f) ? (_Float16)1.0f : (_Float16)0.0f;
    }
    __syncthreads();

    // ---------- conv2 (16->32, k=5) as WMMA: 20 pos x 2 ntiles, K=80 pad 96 ----
    // Membranes in LDS (lane-private packed), loop rolled to bound VGPRs.
#pragma unroll 1
    for (int i = 0; i < 5; ++i) {
      const int tt = w * 5 + i;
      const int p = tt >> 1, nt = tt & 1;
      v8f acc = {};
#pragma unroll
      for (int ks = 0; ks < 3; ++ks) {
        const int c0 = ks * 32 + hi * 8;      // K = tap*16 + ci
        const int c1 = c0 + 16;
        AFrag a;
        a.h[0] = *(const v8h*)&s1[(lm * 25 + p + (c0 >> 4)) * 16 + (c0 & 15)];
        a.h[1] = *(const v8h*)&s1[(lm * 25 + p + (c1 >> 4)) * 16 + (c1 & 15)];
        const v16h bf = *(const v16h*)&wpk[OFF_W2 + ((nt * 3 + ks) * 32 + lane) * 16];
        acc = WMMA_F16(a.v, bf, acc);
      }
      const float bias = blds[nt * 16 + lm];
      float* mp = &m2lds[((tt * 2 + hi) * 16 + lm) * 8];
      MChunk mc;
      mc.v[0] = *(const float4*)mp;
      mc.v[1] = *(const float4*)(mp + 4);
#pragma unroll
      for (int r = 0; r < 8; ++r) {
        const int bb = hi * 8 + r;            // D row -> batch
        float m   = mc.f[r];
        float rst = (m > 1.0f) ? 1.0f : 0.0f;
        m = 0.9f * m + (acc[r] + bias) - rst;
        mc.f[r] = m;
        s2[(bb * 20 + p) * 32 + nt * 16 + lm] = (m > 1.0f) ? (_Float16)1.0f : (_Float16)0.0f;
      }
      *(float4*)mp       = mc.v[0];
      *(float4*)(mp + 4) = mc.v[1];
    }
    __syncthreads();

    // ---------- conv3 (32->64, k=3) as WMMA: 18 pos x 4 ntiles, K=96 exact ----
#pragma unroll 1
    for (int i = 0; i < 9; ++i) {
      const int tt = w * 9 + i;
      const int p = tt >> 2, nt = tt & 3;
      v8f acc = {};
#pragma unroll
      for (int ks = 0; ks < 3; ++ks) {        // kstep == tap
        const _Float16* row = &s2[(lm * 20 + p + ks) * 32 + hi * 8];
        AFrag a;
        a.h[0] = *(const v8h*)row;
        a.h[1] = *(const v8h*)(row + 16);
        const v16h bf = *(const v16h*)&wpk[OFF_W3 + ((nt * 3 + ks) * 32 + lane) * 16];
        acc = WMMA_F16(a.v, bf, acc);
      }
      const float bias = blds[32 + nt * 16 + lm];
      float* mp = &m3lds[((tt * 2 + hi) * 16 + lm) * 8];
      MChunk mc;
      mc.v[0] = *(const float4*)mp;
      mc.v[1] = *(const float4*)(mp + 4);
#pragma unroll
      for (int r = 0; r < 8; ++r) {
        const int bb = hi * 8 + r;
        float m   = mc.f[r];
        float rst = (m > 1.0f) ? 1.0f : 0.0f;
        m = 0.9f * m + (acc[r] + bias) - rst;
        mc.f[r] = m;
        // s3 row layout pos*64+ch matches fw1 prepack K-permutation
        s3[bb * 1152 + p * 64 + nt * 16 + lm] = (m > 1.0f) ? (_Float16)1.0f : (_Float16)0.0f;
      }
      *(float4*)mp       = mc.v[0];
      *(float4*)(mp + 4) = mc.v[1];
    }
    __syncthreads();

    // ---------- fc1 (1152->256) as WMMA: 16 ntiles x 36 ksteps ----------
#pragma unroll
    for (int i = 0; i < 2; ++i) {
      const int nt = w * 2 + i;
      v8f acc = {};
      const _Float16* bp = &wpk[OFF_FW1 + nt * 36 * 512 + lane * 16];
      const _Float16* ap = &s3[lm * 1152 + hi * 8];
      for (int ks = 0; ks < 36; ++ks) {
        AFrag a;
        a.h[0] = *(const v8h*)(ap + ks * 32);
        a.h[1] = *(const v8h*)(ap + ks * 32 + 16);
        const v16h bf = *(const v16h*)(bp + ks * 512);
        acc = WMMA_F16(a.v, bf, acc);
      }
#pragma unroll
      for (int r = 0; r < 8; ++r) {
        const int bb = hi * 8 + r;
        float m   = m4[i][r];
        float rst = (m > 1.0f) ? 1.0f : 0.0f;
        m = 0.9f * m + (acc[r] + bias4[i]) - rst;
        m4[i][r] = m;
        s4[bb * 256 + nt * 16 + lm] = (m > 1.0f) ? (_Float16)1.0f : (_Float16)0.0f;
      }
    }
    __syncthreads();

    // ---------- fc2 (256->10, N padded to 16) : wave 0 ----------
    if (w == 0) {
      v8f acc = {};
#pragma unroll
      for (int ks = 0; ks < 8; ++ks) {
        const _Float16* ap = &s4[lm * 256 + ks * 32 + hi * 8];
        AFrag a;
        a.h[0] = *(const v8h*)ap;
        a.h[1] = *(const v8h*)(ap + 16);
        const v16h bf = *(const v16h*)&wpk[OFF_FW2 + (ks * 32 + lane) * 16];
        acc = WMMA_F16(a.v, bf, acc);
      }
#pragma unroll
      for (int r = 0; r < 8; ++r) {
        const int bb = hi * 8 + r;
        float m   = m5[r];
        float rst = (m > 1.0f) ? 1.0f : 0.0f;
        m = 0.9f * m + (acc[r] + bias5) - rst;
        m5[r] = m;
        if (lm < 10)
          out[((size_t)t * 1024 + bg + bb) * 10 + lm] = (m > 1.0f) ? 1.0f : 0.0f;
      }
    }
    __syncthreads();   // protect s4 (and friends) before next timestep rewrites
  }
}

// ---------------------------------------------------------------------------
// Launcher. Inputs (setup_inputs order):
// 0:x 1:w1 2:b1 3:w2 4:b2 5:w3 6:b3 7:fw1 8:fb1 9:fw2 10:fb2   (all f32)
// d_out: (100,1024,10) f32.  d_ws: >= 616448 bytes used for packed weights.
// ---------------------------------------------------------------------------
extern "C" void kernel_launch(void* const* d_in, const int* in_sizes, int n_in,
                              void* d_out, int out_size, void* d_ws, size_t ws_size,
                              hipStream_t stream) {
  (void)in_sizes; (void)n_in; (void)out_size; (void)ws_size;
  const float* x   = (const float*)d_in[0];
  const float* w1  = (const float*)d_in[1];
  const float* b1  = (const float*)d_in[2];
  const float* w2  = (const float*)d_in[3];
  const float* b2  = (const float*)d_in[4];
  const float* w3  = (const float*)d_in[5];
  const float* b3  = (const float*)d_in[6];
  const float* fw1 = (const float*)d_in[7];
  const float* fb1 = (const float*)d_in[8];
  const float* fw2 = (const float*)d_in[9];
  const float* fb2 = (const float*)d_in[10];
  _Float16* ws = reinterpret_cast<_Float16*>(d_ws);

  pack_fw1<<<1152, 256, 0, stream>>>(fw1, ws);   // 294912 halves
  pack_w2 <<<  12, 256, 0, stream>>>(w2,  ws);   //   3072 halves
  pack_w3 <<<  24, 256, 0, stream>>>(w3,  ws);   //   6144 halves
  pack_fw2<<<  16, 256, 0, stream>>>(fw2, ws);   //   4096 halves

  snn_fused<<<64, 256, 0, stream>>>(x, w1, b1, b2, b3, fb1, fb2, ws, (float*)d_out);
}
